// ParallelDroplessMinions_27255862460666
// MI455X (gfx1250) — matrix operand order repaired and checked by reading, not compiled, not run on hardware
//
#include <hip/hip_runtime.h>
#include <hip/hip_bf16.h>

// ---------------------------------------------------------------------------
// MoE-LoRA grouped FFN for MI455X (gfx1250, wave32, WMMA bf16 16x16x32)
// ---------------------------------------------------------------------------
#define SL 2048
#define BS 4
#define HS 1024
#define FF 4096
#define NE 8
#define RANK 64
#define TOPK 2
#define T_TOK (SL * BS)          // 8192 tokens
#define R_ROWS (T_TOK * TOPK)    // 16384 routed rows
#define CHUNK 512                // FF processed in chunks of 512
#define MAX_TILES (R_ROWS / 16 + NE)   // 1032

typedef __attribute__((ext_vector_type(16))) __bf16 v16bf;
typedef __attribute__((ext_vector_type(8)))  float  v8f;

union BfOp {
    v16bf v;
    uint4 q[2];
};

__device__ __forceinline__ unsigned short f2bf(float f) {
    unsigned u = __float_as_uint(f);
    unsigned r = u + 0x7FFFu + ((u >> 16) & 1u);   // round-to-nearest-even
    return (unsigned short)(r >> 16);
}

__device__ __forceinline__ float gelu_exact(float x) {
    return 0.5f * x * (1.0f + erff(x * 0.70710678118654752f));
}

// ---------------------------------------------------------------------------
// Kernel 1: deterministic expert binning (single block, 8 waves, wave e owns
// expert e).  Produces, in ws header:
//   hdr[0..8]   : row offsets per expert (exclusive scan, hdr[8] == R)
//   hdr[16..24] : tile offsets per expert (16-row tiles, hdr[24] == #tiles)
// plus perm_tok[R] (source token) and perm_w[R] (routing weight).
// ---------------------------------------------------------------------------
__global__ void route_kernel(const float* __restrict__ ew,
                             const int*   __restrict__ ei,
                             int*   __restrict__ hdr,
                             int*   __restrict__ perm_tok,
                             float* __restrict__ perm_w) {
    __shared__ int sCnt[NE];
    __shared__ int sOff[NE + 1];
    const int tid  = threadIdx.x;
    const int wave = tid >> 5;
    const int lane = tid & 31;

    // pass 1: count rows per expert
    int cnt = 0;
    for (int r0 = 0; r0 < R_ROWS; r0 += 32) {
        int r = r0 + lane;
        bool m = (ei[r] == wave);
        unsigned long long mask = __ballot(m);
        cnt += __popcll(mask);
    }
    if (lane == 0) sCnt[wave] = cnt;
    __syncthreads();

    if (tid == 0) {
        int off = 0, toff = 0;
        for (int e = 0; e < NE; ++e) {
            hdr[e]      = off;   sOff[e] = off;
            hdr[16 + e] = toff;
            off  += sCnt[e];
            toff += (sCnt[e] + 15) >> 4;
        }
        hdr[8]  = off;  sOff[NE] = off;
        hdr[24] = toff;
    }
    __syncthreads();

    // pass 2: stable scatter into expert-sorted order
    int base = sOff[wave];
    int done = 0;
    for (int r0 = 0; r0 < R_ROWS; r0 += 32) {
        int r = r0 + lane;
        bool m = (ei[r] == wave);
        unsigned long long mask = __ballot(m);
        int prefix = __popcll(mask & ((1ull << lane) - 1ull));
        if (m) {
            int pos = base + done + prefix;
            perm_tok[pos] = r / TOPK;
            perm_w[pos]   = ew[r];
        }
        done += __popcll(mask);
    }
}

// ---------------------------------------------------------------------------
// Kernel 2: convert weights fp32 -> bf16, pre-transposing w1_B / w2_B to
// N-major [E][N][RANK] so all WMMA B-operands are contiguous-K per lane.
//   w1A_bf : [E*RANK][HS]   (as-is)       w1Bt_bf : [E][FF][RANK]
//   w2A_bf : [E*RANK][FF]   (as-is)       w2Bt_bf : [E][HS][RANK]
// ---------------------------------------------------------------------------
#define N_W1A (NE * RANK * HS)
#define N_W1B (NE * RANK * FF)
#define N_W2A (NE * RANK * FF)
#define N_W2B (NE * RANK * HS)
#define N_TOT (N_W1A + N_W1B + N_W2A + N_W2B)

__global__ void convert_weights_kernel(const float* __restrict__ w1A,
                                       const float* __restrict__ w1B,
                                       const float* __restrict__ w2A,
                                       const float* __restrict__ w2B,
                                       unsigned short* __restrict__ w1A_bf,
                                       unsigned short* __restrict__ w1Bt_bf,
                                       unsigned short* __restrict__ w2A_bf,
                                       unsigned short* __restrict__ w2Bt_bf) {
    long long idx = (long long)blockIdx.x * blockDim.x + threadIdx.x;
    if (idx >= N_TOT) return;
    if (idx < N_W1A) {
        w1A_bf[idx] = f2bf(w1A[idx]);
    } else if (idx < N_W1A + N_W1B) {
        long long o = idx - N_W1A;                // out index in [E][FF][RANK]
        int e = (int)(o / (FF * RANK));
        int rem = (int)(o - (long long)e * (FF * RANK));
        int f = rem / RANK, r = rem % RANK;
        w1Bt_bf[o] = f2bf(w1B[(long long)(e * RANK + r) * FF + f]);
    } else if (idx < N_W1A + N_W1B + N_W2A) {
        long long o = idx - (N_W1A + N_W1B);
        w2A_bf[o] = f2bf(w2A[o]);
    } else {
        long long o = idx - (N_W1A + N_W1B + N_W2A); // out [E][HS][RANK]
        int e = (int)(o / (HS * RANK));
        int rem = (int)(o - (long long)e * (HS * RANK));
        int h = rem / RANK, r = rem % RANK;
        w2Bt_bf[o] = f2bf(w2B[(long long)(e * RANK + r) * HS + h]);
    }
}

// ---------------------------------------------------------------------------
// Kernel 3: main tile kernel.  One workgroup (256 threads = 8 wave32) per
// 16-row tile of expert-sorted routed rows.  All LDS WMMA-operand loads are
// expressed in uint4 units off a 16B-aligned base so they lower to
// ds_load_b128; all global B-operand loads use a hoisted per-wave base with
// constant offsets so they fold into the 24-bit immediate offset field.
// ---------------------------------------------------------------------------
struct __align__(16) Smem {
    unsigned short sX[16 * HS];     // off 0      (32768 B)
    unsigned short sHc[16 * CHUNK]; // off 32768  (16384 B)
    unsigned short sU[16 * RANK];   // off 49152  ( 2048 B)
    unsigned short sV[16 * RANK];   // off 51200  ( 2048 B)
    float sRed[4 * 256];            // off 53248  ( 4096 B)
    int   tok_s[16];
    float w_s[16];
    int   valid_s[16];
    int   sMeta[4];
};

__global__ void __launch_bounds__(256)
moe_tile_kernel(const float* __restrict__ x,
                const int*   __restrict__ hdr,
                const int*   __restrict__ perm_tok,
                const float* __restrict__ perm_w,
                const unsigned short* __restrict__ w1A_bf,
                const unsigned short* __restrict__ w1Bt_bf,
                const unsigned short* __restrict__ w2A_bf,
                const unsigned short* __restrict__ w2Bt_bf,
                float* __restrict__ out) {
    __shared__ Smem sm;
    const int tid  = threadIdx.x;
    const int wave = tid >> 5;
    const int lane = tid & 31;
    const int lm   = lane & 15;     // column / row-in-half selector
    const int lh   = lane >> 4;     // lane half (K offset selector)

    // --- map block -> (expert, row range) ---------------------------------
    if (tid == 0) {
        int ntiles = hdr[24];
        if ((int)blockIdx.x >= ntiles) {
            sm.sMeta[0] = -1;
        } else {
            int e = 0;
            while (e < NE - 1 && (int)blockIdx.x >= hdr[16 + e + 1]) e++;
            int row_start = hdr[e] + ((int)blockIdx.x - hdr[16 + e]) * 16;
            int rows = hdr[e + 1] - row_start;
            if (rows > 16) rows = 16;
            sm.sMeta[0] = e; sm.sMeta[1] = row_start; sm.sMeta[2] = rows;
        }
    }
    __syncthreads();
    if (sm.sMeta[0] < 0) return;
    const int e = sm.sMeta[0];
    const int row_start = sm.sMeta[1];
    const int rows = sm.sMeta[2];

    if (tid < 16) {
        if (tid < rows) {
            int pr = row_start + tid;
            sm.tok_s[tid] = perm_tok[pr];
            sm.w_s[tid]   = perm_w[pr];
            sm.valid_s[tid] = 1;
        } else {
            sm.tok_s[tid] = 0; sm.w_s[tid] = 0.0f; sm.valid_s[tid] = 0;
        }
    }
    __syncthreads();

    // LDS viewed in uint4 (8 bf16) units — provably 16B aligned
    const uint4* qX  = (const uint4*)sm.sX;    // row stride HS/8   = 128
    const uint4* qHc = (const uint4*)sm.sHc;   // row stride CHUNK/8 = 64
    const uint4* qU  = (const uint4*)sm.sU;    // row stride RANK/8  = 8
    const uint4* qV  = (const uint4*)sm.sV;

    // --- stage gathered X tile as bf16 (row-major [16][HS]) ---------------
    for (int i = tid; i < 16 * (HS / 8); i += 256) {   // 2048 uint4 total
        int row = i >> 7;                              // 128 uint4 per row
        int c8  = i & 127;
        const float4* px = (const float4*)(x + (size_t)sm.tok_s[row] * HS) + c8 * 2;
        float4 v0 = px[0], v1 = px[1];
        uint4 b;
        b.x = ((unsigned)f2bf(v0.y) << 16) | f2bf(v0.x);
        b.y = ((unsigned)f2bf(v0.w) << 16) | f2bf(v0.z);
        b.z = ((unsigned)f2bf(v1.y) << 16) | f2bf(v1.x);
        b.w = ((unsigned)f2bf(v1.w) << 16) | f2bf(v1.z);
        ((uint4*)sm.sX)[row * 128 + c8] = b;
    }
    __syncthreads();

    const int ntile = wave & 3;   // which 16-wide rank tile (GEMM1/3)
    const int kh    = wave >> 2;  // split-K half (GEMM1/3)

    // --- GEMM1: U[16x64] = Xtile[16x1024] @ w1A[e]^T ----------------------
    {
        const int aBase = lm * 128 + kh * 64 + 2 * lh;      // uint4 idx into sX
        const int n1 = ntile * 16 + lm;
        const uint4* pB1 = (const uint4*)(w1A_bf + ((e * RANK + n1) << 10)
                                          + kh * 512 + 16 * lh);
        v8f accU = {0.f, 0.f, 0.f, 0.f, 0.f, 0.f, 0.f, 0.f};
        #pragma unroll
        for (int ks = 0; ks < 16; ++ks) {
            BfOp A, B;
            A.q[0] = qX[aBase + ks * 4];
            A.q[1] = qX[aBase + ks * 4 + 1];
            B.q[0] = pB1[ks * 4];
            B.q[1] = pB1[ks * 4 + 1];
            accU = __builtin_amdgcn_wmma_f32_16x16x32_bf16(
                false, A.v, false, B.v, (short)0, accU, false, false);
        }
        if (kh == 1) {
            #pragma unroll
            for (int v = 0; v < 8; ++v) sm.sRed[ntile * 256 + v * 32 + lane] = accU[v];
        }
        __syncthreads();
        if (kh == 0) {
            #pragma unroll
            for (int v = 0; v < 8; ++v) {
                float s = accU[v] + sm.sRed[ntile * 256 + v * 32 + lane];
                sm.sU[(v + 8 * lh) * RANK + ntile * 16 + lm] = f2bf(s);
            }
        }
        __syncthreads();
    }

    // --- hoist GEMM2 A operand (U is invariant over all chunks) -----------
    BfOp A2[2];
    A2[0].q[0] = qU[lm * 8 + 2 * lh];        // k0 = 0
    A2[0].q[1] = qU[lm * 8 + 2 * lh + 1];
    A2[1].q[0] = qU[lm * 8 + 2 * lh + 4];    // k0 = 32
    A2[1].q[1] = qU[lm * 8 + 2 * lh + 5];

    // --- fused GEMM2 (H chunk) + GEMM3 (V partial) over FF chunks ---------
    // per-wave global bases with constant unrolled offsets
    const uint4* pB2w = (const uint4*)(w1Bt_bf + ((e * FF + wave * 64 + lm) << 6)
                                       + 16 * lh);                  // + c*CHUNK rows
    const uint4* pB3w = (const uint4*)(w2A_bf + ((e * RANK + ntile * 16 + lm) << 12)
                                       + kh * 256 + 16 * lh);       // + c*CHUNK cols
    v8f accV = {0.f, 0.f, 0.f, 0.f, 0.f, 0.f, 0.f, 0.f};
    for (int c = 0; c < FF / CHUNK; ++c) {
        const uint4* pB2 = pB2w + (size_t)c * (CHUNK * RANK / 8);   // c*4096 uint4
        const uint4* pB3 = pB3w + (size_t)c * (CHUNK / 8);          // c*64 uint4

        // GEMM2: each wave computes 4 adjacent 16-wide n-tiles of the chunk
        #pragma unroll
        for (int nt = 0; nt < 4; ++nt) {
            v8f acc = {0.f, 0.f, 0.f, 0.f, 0.f, 0.f, 0.f, 0.f};
            #pragma unroll
            for (int ks = 0; ks < 2; ++ks) {
                BfOp B;
                B.q[0] = pB2[nt * 128 + ks * 4];
                B.q[1] = pB2[nt * 128 + ks * 4 + 1];
                acc = __builtin_amdgcn_wmma_f32_16x16x32_bf16(
                    false, A2[ks].v, false, B.v, (short)0, acc, false, false);
            }
            int ncol = (wave * 4 + nt) * 16;     // within chunk
            #pragma unroll
            for (int v = 0; v < 8; ++v) {
                float h = gelu_exact(acc[v]);
                sm.sHc[(v + 8 * lh) * CHUNK + ncol + lm] = f2bf(h);
            }
        }
        __syncthreads();

        // GEMM3 partial: accV += Hchunk[16x512] @ w2A[e][:, chunk]^T
        const int aBase = lm * 64 + kh * 32 + 2 * lh;    // uint4 idx into sHc
        #pragma unroll
        for (int ks = 0; ks < 8; ++ks) {
            BfOp A, B;
            A.q[0] = qHc[aBase + ks * 4];
            A.q[1] = qHc[aBase + ks * 4 + 1];
            B.q[0] = pB3[ks * 4];
            B.q[1] = pB3[ks * 4 + 1];
            accV = __builtin_amdgcn_wmma_f32_16x16x32_bf16(
                false, A.v, false, B.v, (short)0, accV, false, false);
        }
        __syncthreads();   // before next chunk overwrites sHc
    }

    // --- reduce split-K V and store bf16 ----------------------------------
    if (kh == 1) {
        #pragma unroll
        for (int v = 0; v < 8; ++v) sm.sRed[ntile * 256 + v * 32 + lane] = accV[v];
    }
    __syncthreads();
    if (kh == 0) {
        #pragma unroll
        for (int v = 0; v < 8; ++v) {
            float s = accV[v] + sm.sRed[ntile * 256 + v * 32 + lane];
            sm.sV[(v + 8 * lh) * RANK + ntile * 16 + lm] = f2bf(s);
        }
    }
    __syncthreads();

    // --- GEMM4: Y[16x1024] = V[16x64] @ w2B[e]; weighted scatter ----------
    BfOp A4[2];
    A4[0].q[0] = qV[lm * 8 + 2 * lh];
    A4[0].q[1] = qV[lm * 8 + 2 * lh + 1];
    A4[1].q[0] = qV[lm * 8 + 2 * lh + 4];
    A4[1].q[1] = qV[lm * 8 + 2 * lh + 5];

    const uint4* pB4 = (const uint4*)(w2Bt_bf + ((e * HS + wave * 128 + lm) << 6)
                                      + 16 * lh);
    #pragma unroll
    for (int nt = 0; nt < 8; ++nt) {
        v8f acc = {0.f, 0.f, 0.f, 0.f, 0.f, 0.f, 0.f, 0.f};
        #pragma unroll
        for (int ks = 0; ks < 2; ++ks) {
            BfOp B;
            B.q[0] = pB4[nt * 128 + ks * 4];
            B.q[1] = pB4[nt * 128 + ks * 4 + 1];
            acc = __builtin_amdgcn_wmma_f32_16x16x32_bf16(
                false, A4[ks].v, false, B.v, (short)0, acc, false, false);
        }
        int n0 = (wave * 8 + nt) * 16;
        #pragma unroll
        for (int v = 0; v < 8; ++v) {
            int m = v + 8 * lh;
            if (sm.valid_s[m]) {
                atomicAdd(&out[(size_t)sm.tok_s[m] * HS + n0 + lm],
                          sm.w_s[m] * acc[v]);
            }
        }
    }
}

// ---------------------------------------------------------------------------
// Host launch
// ---------------------------------------------------------------------------
extern "C" void kernel_launch(void* const* d_in, const int* in_sizes, int n_in,
                              void* d_out, int out_size, void* d_ws, size_t ws_size,
                              hipStream_t stream) {
    const float* x   = (const float*)d_in[0];
    const float* ew  = (const float*)d_in[1];
    const int*   ei  = (const int*)d_in[2];
    const float* w1A = (const float*)d_in[3];
    const float* w1B = (const float*)d_in[4];
    const float* w2A = (const float*)d_in[5];
    const float* w2B = (const float*)d_in[6];
    float* out = (float*)d_out;

    char* ws = (char*)d_ws;
    size_t off = 0;
    int*   hdr       = (int*)(ws + off);            off += 256;
    int*   perm_tok  = (int*)(ws + off);            off += (size_t)R_ROWS * 4;
    float* perm_w    = (float*)(ws + off);          off += (size_t)R_ROWS * 4;
    unsigned short* w1A_bf  = (unsigned short*)(ws + off); off += (size_t)N_W1A * 2;
    unsigned short* w1Bt_bf = (unsigned short*)(ws + off); off += (size_t)N_W1B * 2;
    unsigned short* w2A_bf  = (unsigned short*)(ws + off); off += (size_t)N_W2A * 2;
    unsigned short* w2Bt_bf = (unsigned short*)(ws + off); off += (size_t)N_W2B * 2;

    hipMemsetAsync(d_out, 0, (size_t)T_TOK * HS * sizeof(float), stream);

    route_kernel<<<1, 256, 0, stream>>>(ew, ei, hdr, perm_tok, perm_w);

    convert_weights_kernel<<<(N_TOT + 255) / 256, 256, 0, stream>>>(
        w1A, w1B, w2A, w2B, w1A_bf, w1Bt_bf, w2A_bf, w2Bt_bf);

    moe_tile_kernel<<<MAX_TILES, 256, 0, stream>>>(
        x, hdr, perm_tok, perm_w, w1A_bf, w1Bt_bf, w2A_bf, w2Bt_bf, out);
}
